// ContactPredictionHead_68221260530028
// MI455X (gfx1250) — matrix operand order repaired and checked by reading, not compiled, not run on hardware
//
#include <hip/hip_runtime.h>
#include <hip/hip_bf16.h>

// Problem constants (from the reference file)
#define B_  4
#define L_  2048
#define D_  1280
#define O_  2

// Tiling
#define TILE       128           // output tile is 128x128 (both o's)
#define KB         32            // K per WMMA step (bf16 16x16x32)
#define NSTEPS     (D_ / KB)     // 40
#define LDS_STRIDE 40            // 32 + 8 pad bf16 elems per row -> 80B row stride (bank-conflict free)
#define NTILES     (L_ / TILE)   // 16
#define NPAIRS     (NTILES * (NTILES + 1) / 2)  // 136 upper-triangle tile pairs

typedef __attribute__((ext_vector_type(16))) __bf16 v16bf;
typedef __attribute__((ext_vector_type(8)))  __bf16 v8bf;
typedef __attribute__((ext_vector_type(8)))  float  v8f;
typedef __attribute__((ext_vector_type(2)))  float  v2f;

static __device__ inline v16bf cat16(v8bf lo, v8bf hi) {
    v16bf r;
#pragma unroll
    for (int i = 0; i < 8; ++i) { r[i] = lo[i]; r[i + 8] = hi[i]; }
    return r;
}

// pack 4 scaled fp32 -> 4 bf16, one 8B LDS store
static __device__ inline void store4bf_scaled(__bf16* p, float4 v, float4 w) {
    union { __bf16 h[4]; unsigned long long u; } pk;
    pk.h[0] = (__bf16)(v.x * w.x);
    pk.h[1] = (__bf16)(v.y * w.y);
    pk.h[2] = (__bf16)(v.z * w.z);
    pk.h[3] = (__bf16)(v.w * w.w);
    *(unsigned long long*)p = pk.u;
}
static __device__ inline void store4bf(__bf16* p, float4 v) {
    union { __bf16 h[4]; unsigned long long u; } pk;
    pk.h[0] = (__bf16)v.x;
    pk.h[1] = (__bf16)v.y;
    pk.h[2] = (__bf16)v.z;
    pk.h[3] = (__bf16)v.w;
    *(unsigned long long*)p = pk.u;
}

// out[b,i,j,o] = sum_d h[b,i,d] h[b,j,d] Wp[o,d] + bias[o]   (symmetric in i,j)
__global__ __launch_bounds__(256)
void contact_head_gemm(const float* __restrict__ H,   // (B, L, D)
                       const float* __restrict__ W,   // (O, 2D); Wp = W[:, :D]
                       const float* __restrict__ bias,// (O,)
                       float* __restrict__ out)       // (B, L, L, O)
{
    // double-buffered bf16 staging tiles
    __shared__ __align__(16) __bf16 ldsA0[2][TILE * LDS_STRIDE]; // h_i * Wp[0]
    __shared__ __align__(16) __bf16 ldsA1[2][TILE * LDS_STRIDE]; // h_i * Wp[1]
    __shared__ __align__(16) __bf16 ldsB [2][TILE * LDS_STRIDE]; // h_j

    const int batch = blockIdx.y;

    // map blockIdx.x -> (ti, tj) with ti <= tj over 16x16 tile grid
    int rem = blockIdx.x;
    int ti = 0, rowlen = NTILES;
    while (rem >= rowlen) { rem -= rowlen; ++ti; --rowlen; }
    const int tj = ti + rem;
    const int i0 = ti * TILE;
    const int j0 = tj * TILE;

    const int tid  = threadIdx.x;
    const int lane = tid & 31;
    const int wave = tid >> 5;
    const int rowbase = (wave & 3) * 32;   // wave's 32-row slice of the 128 rows
    const int colbase = (wave >> 2) * 64;  // wave's 64-col slice of the 128 cols

    const int lrow = lane & 15;
    const int hi16 = lane >> 4;            // which half-wave

    // staging assignment: tid&7 picks the float4 column (same for all q), tid>>3 picks base row
    const int c4   = (tid & 7) * 4;        // column (floats) 0,4,...,28 — invariant per thread
    const int row0 = tid >> 3;             // rows row0 + 32*q, q = 0..3

    const size_t hbase = (size_t)batch * L_ * D_;
    const float* Hi = H + hbase + (size_t)i0 * D_ + c4;
    const float* Hj = H + hbase + (size_t)j0 * D_ + c4;

    v8f acc[2][2][4];                      // [o][r(16-row tile)][c(16-col tile)]
#pragma unroll
    for (int o = 0; o < 2; ++o)
#pragma unroll
        for (int r = 0; r < 2; ++r)
#pragma unroll
            for (int c = 0; c < 4; ++c)
#pragma unroll
                for (int e = 0; e < 8; ++e) acc[o][r][c][e] = 0.0f;

    const float b0 = bias[0];
    const float b1 = bias[1];

    // register staging for one K-step of global data
    float4 rhi[4], rhj[4], rw0, rw1;

    // ---- prologue: fetch + stage step 0 into buffer 0 ----
    {
        const int k0 = 0;
        rw0 = *(const float4*)(W +          k0 + c4);
        rw1 = *(const float4*)(W + 2 * D_ + k0 + c4);
#pragma unroll
        for (int q = 0; q < 4; ++q) {
            rhi[q] = *(const float4*)(Hi + (size_t)(row0 + 32 * q) * D_ + k0);
            rhj[q] = *(const float4*)(Hj + (size_t)(row0 + 32 * q) * D_ + k0);
        }
#pragma unroll
        for (int q = 0; q < 4; ++q) {
            const int row = row0 + 32 * q;
            store4bf_scaled(ldsA0[0] + row * LDS_STRIDE + c4, rhi[q], rw0);
            store4bf_scaled(ldsA1[0] + row * LDS_STRIDE + c4, rhi[q], rw1);
            store4bf       (ldsB [0] + row * LDS_STRIDE + c4, rhj[q]);
        }
    }
    __syncthreads();

    int buf = 0;
    for (int step = 0; step < NSTEPS; ++step) {
        const bool has_next = (step + 1 < NSTEPS);

        // ---- issue next step's global loads first (latency hidden under WMMA) ----
        if (has_next) {
            const int k0 = (step + 1) * KB;
            rw0 = *(const float4*)(W +          k0 + c4);
            rw1 = *(const float4*)(W + 2 * D_ + k0 + c4);
#pragma unroll
            for (int q = 0; q < 4; ++q) {
                rhi[q] = *(const float4*)(Hi + (size_t)(row0 + 32 * q) * D_ + k0);
                rhj[q] = *(const float4*)(Hj + (size_t)(row0 + 32 * q) * D_ + k0);
            }
        }

        // ---- fragments (ISA 16-bit wave32 layouts) + 16 WMMA ----
        const __bf16* A0 = ldsA0[buf];
        const __bf16* A1 = ldsA1[buf];
        const __bf16* Bt = ldsB [buf];

        // A: lane holds row (lane&15); lanes<16 -> K {0..7,16..23}, lanes>=16 -> K {8..15,24..31}
        v16bf afrag[2][2];
#pragma unroll
        for (int r = 0; r < 2; ++r) {
            const int arow = rowbase + r * 16 + lrow;
            const __bf16* pa0 = A0 + arow * LDS_STRIDE + hi16 * 8;
            const __bf16* pa1 = A1 + arow * LDS_STRIDE + hi16 * 8;
            afrag[0][r] = cat16(*(const v8bf*)pa0, *(const v8bf*)(pa0 + 16));
            afrag[1][r] = cat16(*(const v8bf*)pa1, *(const v8bf*)(pa1 + 16));
        }
        // B = H^T: lane holds column (lane&15) = H row j; lanes<16 -> K {0..7,16..23}, lanes>=16 -> K {8..15,24..31}
        v16bf bfrag[4];
#pragma unroll
        for (int c = 0; c < 4; ++c) {
            const __bf16* pb = Bt + (colbase + c * 16 + lrow) * LDS_STRIDE + hi16 * 8;
            bfrag[c] = cat16(*(const v8bf*)pb, *(const v8bf*)(pb + 16));
        }

#pragma unroll
        for (int r = 0; r < 2; ++r)
#pragma unroll
            for (int c = 0; c < 4; ++c) {
                acc[0][r][c] = __builtin_amdgcn_wmma_f32_16x16x32_bf16(
                    false, afrag[0][r], false, bfrag[c], (short)0, acc[0][r][c], false, false);
                acc[1][r][c] = __builtin_amdgcn_wmma_f32_16x16x32_bf16(
                    false, afrag[1][r], false, bfrag[c], (short)0, acc[1][r][c], false, false);
            }

        // ---- convert + store next step into the other buffer, single barrier ----
        if (has_next) {
            const int nb = buf ^ 1;
#pragma unroll
            for (int q = 0; q < 4; ++q) {
                const int row = row0 + 32 * q;
                store4bf_scaled(ldsA0[nb] + row * LDS_STRIDE + c4, rhi[q], rw0);
                store4bf_scaled(ldsA1[nb] + row * LDS_STRIDE + c4, rhi[q], rw1);
                store4bf       (ldsB [nb] + row * LDS_STRIDE + c4, rhj[q]);
            }
            __syncthreads();
            buf = nb;
        }
    }

    // ---- store: C layout is VGPR e -> M = e + 8*hi16, N = lane&15 ----
    float* outb = out + (size_t)batch * L_ * L_ * O_;
    const bool mirror = (ti != tj);
#pragma unroll
    for (int r = 0; r < 2; ++r) {
        const int igb = i0 + rowbase + r * 16 + hi16 * 8;
#pragma unroll
        for (int c = 0; c < 4; ++c) {
            const int jg = j0 + colbase + c * 16 + lrow;
#pragma unroll
            for (int e = 0; e < 8; ++e) {
                const int ig = igb + e;
                v2f v;
                v.x = acc[0][r][c][e] + b0;
                v.y = acc[1][r][c][e] + b1;
                *(v2f*)(outb + ((size_t)ig * L_ + jg) * O_) = v;      // (i,j,:) coalesced
                if (mirror)
                    *(v2f*)(outb + ((size_t)jg * L_ + ig) * O_) = v;  // symmetric mirror
            }
        }
    }
}

extern "C" void kernel_launch(void* const* d_in, const int* in_sizes, int n_in,
                              void* d_out, int out_size, void* d_ws, size_t ws_size,
                              hipStream_t stream) {
    (void)in_sizes; (void)n_in; (void)out_size; (void)d_ws; (void)ws_size;
    const float* H    = (const float*)d_in[0];
    const float* W    = (const float*)d_in[1];
    const float* bias = (const float*)d_in[2];
    float* out        = (float*)d_out;

    dim3 grid(NPAIRS, B_);
    contact_head_gemm<<<grid, 256, 0, stream>>>(H, W, bias, out);
}
